// Anchor_20444044329185
// MI455X (gfx1250) — matrix operand order repaired and checked by reading, not compiled, not run on hardware
//
#include <hip/hip_runtime.h>
#include <hip/hip_bf16.h>

#define B_ 32
#define N_ 4096
#define D_ 512
#define C_ 64

typedef __bf16 v16bf __attribute__((ext_vector_type(16)));
typedef float  v8f   __attribute__((ext_vector_type(8)));
typedef unsigned short u16v16 __attribute__((ext_vector_type(16)));

__device__ __forceinline__ unsigned short f2bf(float f) {
    // round-to-nearest-even f32 -> bf16
    unsigned int u = __float_as_uint(f);
    u += 0x7fffu + ((u >> 16) & 1u);
    return (unsigned short)(u >> 16);
}

// ---------------------------------------------------------------- kernel 0
__global__ void k_zero(float* __restrict__ out, unsigned int* __restrict__ counts) {
    int i = blockIdx.x * blockDim.x + threadIdx.x;
    if (i < B_ * D_) out[i] = 0.0f;
    if (i < B_ * C_) counts[i] = 0u;
}

// ---------------------------------------------------------------- anchors prep
__global__ void k_prep_anchors(const float* __restrict__ anchors,
                               unsigned short* __restrict__ abf,
                               float* __restrict__ an2) {
    const int c = blockIdx.x;
    const int t = threadIdx.x;            // 64 threads
    float s = 0.0f;
    for (int j = t; j < D_; j += 64) {
        float v = anchors[c * D_ + j];
        abf[c * D_ + j] = f2bf(v);
        s += v * v;
    }
    __shared__ float red[64];
    red[t] = s;
    __syncthreads();
    if (t == 0) {
        float tot = 0.0f;
        for (int k = 0; k < 64; ++k) tot += red[k];
        an2[c] = tot;
    }
}

// ---------------------------------------------------------------- pass 1: WMMA scores
// block = 256 threads = 8 waves; wave handles 16 rows x all 64 clusters.
__global__ void k_scores(const float* __restrict__ x,
                         const long long* __restrict__ mask,
                         const unsigned short* __restrict__ abf,
                         const float* __restrict__ an2,
                         float* __restrict__ score,
                         float* __restrict__ norms,
                         unsigned int* __restrict__ counts) {
    const int b    = blockIdx.y;
    const int tile = blockIdx.x;
    const int wave = threadIdx.x >> 5;
    const int lane = threadIdx.x & 31;
    const int hi   = lane >> 4;
    const int l15  = lane & 15;
    const int row_base = tile * 128 + wave * 16;
    const long long maskb = mask[b];

    // A-fragment row for this lane (both half-waves cover the same 16 rows).
    const float* xrow = x + ((size_t)b * N_ + row_base + l15) * D_;

    v8f vz = {0.f,0.f,0.f,0.f,0.f,0.f,0.f,0.f};
    v8f acc[4];
    #pragma unroll
    for (int t = 0; t < 4; ++t) acc[t] = vz;

    float sumsq = 0.0f;
    const int s0 = hi * 16;   // A-frag K sub-offset (contiguous, matches B)
    const int ks = hi * 16;   // B-frag K sub-offset (contiguous, matches A)

    for (int kc = 0; kc < D_; kc += 32) {
        const float4* p0 = reinterpret_cast<const float4*>(xrow + kc + s0);
        const float4* p1 = reinterpret_cast<const float4*>(xrow + kc + 16 + s0);
        float4 a0 = p0[0], a1 = p0[1];
        float4 b0 = p0[2], b1 = p0[3];
        float f[16] = {a0.x,a0.y,a0.z,a0.w, a1.x,a1.y,a1.z,a1.w,
                       b0.x,b0.y,b0.z,b0.w, b1.x,b1.y,b1.z,b1.w};
        u16v16 ua;
        #pragma unroll
        for (int j = 0; j < 16; ++j) {
            sumsq += f[j] * f[j];
            ua[j] = f2bf(f[j]);
        }
        v16bf afrag = __builtin_bit_cast(v16bf, ua);
        #pragma unroll
        for (int t = 0; t < 4; ++t) {
            const int col = t * 16 + l15;
            const u16v16* pb =
                reinterpret_cast<const u16v16*>(abf + (size_t)col * D_ + kc + ks);
            v16bf bfrag = __builtin_bit_cast(v16bf, pb[0]);
            acc[t] = __builtin_amdgcn_wmma_f32_16x16x32_bf16(
                false, afrag, false, bfrag, (short)0, acc[t], false, false);
        }
    }

    // combine the two half-wave K partitions -> full row sum of squares
    sumsq += __shfl_xor(sumsq, 16, 32);
    float rn = fmaxf(sqrtf(sumsq), 1e-12f);
    float xq = sumsq / (rn * rn);          // == sum(xn*xn)

    __shared__ float sh_r[128];
    __shared__ float sh_q[128];
    if (hi == 0) {
        sh_r[wave * 16 + l15] = rn;
        sh_q[wave * 16 + l15] = xq;
        norms[(size_t)b * N_ + row_base + l15] = rn;
    }
    __syncthreads();

    // accumulator layout: VGPR v holds row (v + hi*8), column = t*16 + l15
    float rr[8], qq[8];
    #pragma unroll
    for (int v = 0; v < 8; ++v) {
        rr[v] = sh_r[wave * 16 + hi * 8 + v];
        qq[v] = sh_q[wave * 16 + hi * 8 + v];
    }
    float a2[4];
    #pragma unroll
    for (int t = 0; t < 4; ++t) a2[t] = an2[t * 16 + l15];

    float inv[4][8];
    #pragma unroll
    for (int t = 0; t < 4; ++t)
        #pragma unroll
        for (int v = 0; v < 8; ++v) {
            float g = acc[t][v] / rr[v];                     // xn . anchor_c
            float sq = qq[v] + a2[t] - 2.0f * g;
            float dist = sqrtf(fmaxf(sq, 0.0f));
            inv[t][v] = 1.0f / dist;
        }

    // row max over C (local over t, butterfly over 16-lane half)
    float m[8];
    #pragma unroll
    for (int v = 0; v < 8; ++v) {
        float mm = fmaxf(fmaxf(inv[0][v], inv[1][v]), fmaxf(inv[2][v], inv[3][v]));
        mm = fmaxf(mm, __shfl_xor(mm, 8, 32));
        mm = fmaxf(mm, __shfl_xor(mm, 4, 32));
        mm = fmaxf(mm, __shfl_xor(mm, 2, 32));
        mm = fmaxf(mm, __shfl_xor(mm, 1, 32));
        m[v] = mm;
    }

    // first-max argmax = min cluster index achieving the max
    int amin[8];
    #pragma unroll
    for (int v = 0; v < 8; ++v) {
        int cand = 1 << 30;
        #pragma unroll
        for (int t = 0; t < 4; ++t)
            if (inv[t][v] == m[v]) cand = min(cand, t * 16 + l15);
        cand = min(cand, __shfl_xor(cand, 8, 32));
        cand = min(cand, __shfl_xor(cand, 4, 32));
        cand = min(cand, __shfl_xor(cand, 2, 32));
        cand = min(cand, __shfl_xor(cand, 1, 32));
        amin[v] = cand;
    }

    // exp + row sum (softmax denominator)
    float s[8];
    #pragma unroll
    for (int v = 0; v < 8; ++v) {
        float ss = 0.0f;
        #pragma unroll
        for (int t = 0; t < 4; ++t) {
            inv[t][v] = __expf(inv[t][v] - m[v]);
            ss += inv[t][v];
        }
        ss += __shfl_xor(ss, 8, 32);
        ss += __shfl_xor(ss, 4, 32);
        ss += __shfl_xor(ss, 2, 32);
        ss += __shfl_xor(ss, 1, 32);
        s[v] = ss;
    }

    // write scores (0 for invalid rows) + vote counts
    #pragma unroll
    for (int v = 0; v < 8; ++v) {
        const int row = hi * 8 + v;
        const int n = row_base + row;
        const bool vld = ((long long)n < maskb);
        const float rs = vld ? (1.0f / s[v]) : 0.0f;
        float* srow = score + ((size_t)b * N_ + n) * C_;
        #pragma unroll
        for (int t = 0; t < 4; ++t)
            srow[t * 16 + l15] = inv[t][v] * rs;
        if (l15 == 0 && vld)
            atomicAdd(&counts[b * C_ + amin[v]], 1u);
    }
}

// ---------------------------------------------------------------- pass 2: mode cluster
__global__ void k_mode(const unsigned int* __restrict__ counts, int* __restrict__ index) {
    int b = threadIdx.x;
    if (b < B_) {
        unsigned int best = counts[b * C_];
        int bi = 0;
        for (int c = 1; c < C_; ++c) {
            unsigned int v = counts[b * C_ + c];
            if (v > best) { best = v; bi = c; }   // strict > -> first max
        }
        index[b] = bi;
    }
}

// ---------------------------------------------------------------- pass 3: weighted sum
__global__ void k_feature(const float* __restrict__ x,
                          const float* __restrict__ score,
                          const float* __restrict__ norms,
                          const int* __restrict__ index,
                          float* __restrict__ out) {
    const int b = blockIdx.y;
    const int n0 = blockIdx.x * 256;
    const int tid = threadIdx.x;          // 256 threads, D = 512
    const int idx = index[b];
    float a0 = 0.0f, a1 = 0.0f;
    for (int r = 0; r < 256; ++r) {
        const int n = n0 + r;
        const float w = score[((size_t)b * N_ + n) * C_ + idx];
        if (w != 0.0f) {
            const float coef = w / norms[(size_t)b * N_ + n];
            const float* xr = x + ((size_t)b * N_ + n) * D_;
            a0 += xr[tid] * coef;
            a1 += xr[tid + 256] * coef;
        }
    }
    atomicAdd(&out[b * D_ + tid], a0);
    atomicAdd(&out[b * D_ + tid + 256], a1);
}

// ---------------------------------------------------------------- launcher
extern "C" void kernel_launch(void* const* d_in, const int* in_sizes, int n_in,
                              void* d_out, int out_size, void* d_ws, size_t ws_size,
                              hipStream_t stream) {
    const float*     x       = (const float*)d_in[0];
    const long long* mask    = (const long long*)d_in[1];   // int64 per reference
    const float*     anchors = (const float*)d_in[2];
    float*           out     = (float*)d_out;

    char* ws = (char*)d_ws;
    const size_t score_bytes = (size_t)B_ * N_ * C_ * sizeof(float);   // 32 MB
    const size_t norms_bytes = (size_t)B_ * N_ * sizeof(float);        // 512 KB
    const size_t abf_bytes   = (size_t)C_ * D_ * sizeof(unsigned short);

    float*          score  = (float*)ws;
    float*          norms  = (float*)(ws + score_bytes);
    unsigned short* abf    = (unsigned short*)(ws + score_bytes + norms_bytes);
    float*          an2    = (float*)((char*)abf + abf_bytes);
    unsigned int*   counts = (unsigned int*)((char*)an2 + C_ * sizeof(float));
    int*            index  = (int*)((char*)counts + B_ * C_ * sizeof(unsigned int));

    k_zero<<<64, 256, 0, stream>>>(out, counts);
    k_prep_anchors<<<C_, 64, 0, stream>>>(anchors, abf, an2);

    dim3 g1(N_ / 128, B_);
    k_scores<<<g1, 256, 0, stream>>>(x, mask, abf, an2, score, norms, counts);

    k_mode<<<1, 32, 0, stream>>>(counts, index);

    dim3 g3(N_ / 256, B_);
    k_feature<<<g3, 256, 0, stream>>>(x, score, norms, index, out);
}
